// task_attention_74010876444748
// MI455X (gfx1250) — compile-verified, matrix operand order checked by reading, not run on hardware
//
#include <hip/hip_runtime.h>

#define B_ 16
#define N_ 16384
#define M_ 256
#define D_ 256

typedef __attribute__((ext_vector_type(16))) __bf16 v16bf;
typedef __attribute__((ext_vector_type(8)))  float  v8f;
typedef unsigned int u32x4 __attribute__((ext_vector_type(4)));
typedef int i32x8 __attribute__((ext_vector_type(8)));
typedef int i32x4 __attribute__((ext_vector_type(4)));

typedef __attribute__((address_space(3))) void as3_void;
typedef __attribute__((address_space(1))) i32x4 gi32x4;  // global int4
typedef __attribute__((address_space(3))) i32x4 li32x4;  // LDS int4

// ---- feature detection (compile-only probing) -------------------------------
#if __has_builtin(__builtin_amdgcn_tensor_load_to_lds) && (__clang_major__ < 23)
#define USE_TDM 1
#endif
#if __has_builtin(__builtin_amdgcn_global_load_async_to_lds_b128)
#define USE_ASYNC 1
#endif

union Frag { v16bf v; __bf16 e[16]; uint4 q4[2]; };
union Acc  { v8f v; float e[8]; };

static __device__ __forceinline__ v8f wmma_bf16(v16bf a, v16bf b, v8f c) {
  return __builtin_amdgcn_wmma_f32_16x16x32_bf16(false, a, false, b, (short)0, c,
                                                 false, false);
}

static __device__ __forceinline__ void cvt8(__bf16* dst, float4 a, float4 b) {
  dst[0] = (__bf16)a.x; dst[1] = (__bf16)a.y; dst[2] = (__bf16)a.z; dst[3] = (__bf16)a.w;
  dst[4] = (__bf16)b.x; dst[5] = (__bf16)b.y; dst[6] = (__bf16)b.z; dst[7] = (__bf16)b.w;
}

// A/B fragment (16x32 bf16, K-major per lane) from an f32 row (converted).
// Element e of lane (r, hi): K = (e<8?0:16) + hi*8 + (e&7)   [CDNA5 ISA 7.12.2]
static __device__ __forceinline__ Frag load_frag_f32(const float* row, int kb, int hi) {
  Frag f;
  float4 a0 = *(const float4*)(row + kb + hi * 8);
  float4 a1 = *(const float4*)(row + kb + hi * 8 + 4);
  float4 a2 = *(const float4*)(row + kb + 16 + hi * 8);
  float4 a3 = *(const float4*)(row + kb + 16 + hi * 8 + 4);
  cvt8(f.e, a0, a1);
  cvt8(f.e + 8, a2, a3);
  return f;
}

// Same fragment pattern from bf16 memory (LDS or global): two 16-byte chunks.
static __device__ __forceinline__ Frag load_frag_bf16(const __bf16* row, int kb, int hi) {
  Frag f;
  f.q4[0] = *(const uint4*)(row + kb + hi * 8);
  f.q4[1] = *(const uint4*)(row + kb + 16 + hi * 8);
  return f;
}

static __device__ __forceinline__ void zero_acc(Acc& a) {
#pragma unroll
  for (int i = 0; i < 8; ++i) a.e[i] = 0.0f;
}

static __device__ __forceinline__ unsigned lds_off(const void* p) {
  return (unsigned)(uintptr_t)(as3_void*)(void*)p;
}

#ifdef USE_ASYNC
static __device__ __forceinline__ void async_cp16(void* lds, const void* g) {
  __builtin_amdgcn_global_load_async_to_lds_b128((gi32x4*)(void*)g, (li32x4*)lds, 0, 0);
}
static __device__ __forceinline__ void wait_async0() {
#if __has_builtin(__builtin_amdgcn_s_wait_asynccnt)
  __builtin_amdgcn_s_wait_asynccnt(0);
#else
  asm volatile("s_wait_asynccnt 0" ::: "memory");
#endif
}
#endif

#ifdef USE_TDM
// Issue a 2D TDM tile load: rows x (w_units * 8B), row stride = stride_units * 8B.
// Descriptor bit layout per CDNA5 ISA 8.3/8.4 (group0: count|lds|global|type,
// group1: data_size=8B, tensor_dim0/1, tile_dim0/1, tensor_dim0_stride).
static __device__ __forceinline__ void tdm_load_2d(unsigned lds_addr, const void* gptr,
                                                   unsigned w_units, unsigned rows,
                                                   unsigned stride_units) {
  unsigned long long ga = (unsigned long long)(uintptr_t)gptr;
  u32x4 g0 = {0, 0, 0, 0};
  g0[0] = 1u;                                                   // count=1 (valid)
  g0[1] = lds_addr;                                             // lds_addr[31:0]
  g0[2] = (unsigned)(ga & 0xffffffffu);                         // global_addr lo
  g0[3] = (unsigned)((ga >> 32) & 0x01ffffffu) | (2u << 30);    // addr hi | type=2
  i32x8 g1 = {0, 0, 0, 0, 0, 0, 0, 0};
  g1[0] = (int)(3u << 16);                                      // data_size = 8B
  g1[1] = (int)((w_units & 0xffffu) << 16);                     // tensor_dim0 lo16
  g1[2] = (int)(((w_units >> 16) & 0xffffu) | ((rows & 0xffffu) << 16)); // dim0hi|dim1lo
  g1[3] = (int)(((rows >> 16) & 0xffffu) | ((w_units & 0xffffu) << 16)); // dim1hi|tile0
  g1[4] = (int)(rows & 0xffffu);                                // tile_dim1 (tile2=0)
  g1[5] = (int)stride_units;                                    // tensor_dim0_stride
  i32x4 z4 = {0, 0, 0, 0};
  __builtin_amdgcn_tensor_load_to_lds(g0, g1, z4, z4, 0);
}
#endif

// ---------------------------------------------------------------- W -> bf16
__global__ void wcvt_kernel(const float* __restrict__ w, __bf16* __restrict__ o) {
  int i = (blockIdx.x * 256 + threadIdx.x) * 4;
  float4 f = *(const float4*)(w + i);
  union { __bf16 e[4]; uint2 u; } p;
  p.e[0] = (__bf16)f.x; p.e[1] = (__bf16)f.y; p.e[2] = (__bf16)f.z; p.e[3] = (__bf16)f.w;
  *(uint2*)(o + i) = p.u;
}

// ------------------------------------------------- k = task@Wk^T+bk, v likewise
// k stored row-major [B][M][D] bf16; v stored transposed [B][D][M] bf16.
__global__ __launch_bounds__(256, 1) void project_kv_kernel(
    const float* __restrict__ task, const __bf16* __restrict__ wkb,
    const float* __restrict__ bk, const __bf16* __restrict__ wvb,
    const float* __restrict__ bv, __bf16* __restrict__ kbf,
    __bf16* __restrict__ vtb) {
  const int t = threadIdx.x, w = t >> 5, lane = t & 31, c = lane & 15, hi = lane >> 4;
  const int b = blockIdx.x >> 1;
  const int m0 = (blockIdx.x & 1) * 128 + w * 16;

  const float* arow = task + ((size_t)b * M_ + m0 + c) * D_;
  Frag afr[8];
#pragma unroll
  for (int kk = 0; kk < 8; ++kk) afr[kk] = load_frag_f32(arow, kk * 32, hi);

#pragma unroll
  for (int ct = 0; ct < 16; ++ct) {
    Acc ka, va;
    zero_acc(ka); zero_acc(va);
    const __bf16* kr = wkb + (size_t)(ct * 16 + c) * D_;
    const __bf16* vr = wvb + (size_t)(ct * 16 + c) * D_;
#pragma unroll
    for (int kk = 0; kk < 8; ++kk) {
      Frag bk_ = load_frag_bf16(kr, kk * 32, hi);
      ka.v = wmma_bf16(afr[kk].v, bk_.v, ka.v);
      Frag bv_ = load_frag_bf16(vr, kk * 32, hi);
      va.v = wmma_bf16(afr[kk].v, bv_.v, va.v);
    }
    float kbias = bk[ct * 16 + c];
    float vbias = bv[ct * 16 + c];
#pragma unroll
    for (int i = 0; i < 8; ++i) {
      int m = m0 + i + 8 * hi;
      int d = ct * 16 + c;
      kbf[((size_t)b * M_ + m) * D_ + d] = (__bf16)(ka.e[i] + kbias);
      vtb[((size_t)b * D_ + d) * M_ + m] = (__bf16)(va.e[i] + vbias);
    }
  }
}

// ---------------------------------------------------------------- fused main
// out = sigmoid((image@Wq^T+bq) @ k^T / 16) @ v + image
__global__ __launch_bounds__(256, 1) void fused_kernel(
    const float* __restrict__ image, const __bf16* __restrict__ Wqb,
    const float* __restrict__ bq, const __bf16* __restrict__ kbf,
    const __bf16* __restrict__ vtb, float* __restrict__ out) {
  __shared__ __align__(16) unsigned char smem[200 * 1024];
  __bf16* wq_lds = (__bf16*)smem;               // [256][256]  (phase A)
  __bf16* k_lds  = (__bf16*)smem;               // [128][256]  (phase B, reuses wq)
  __bf16* vt_lds = (__bf16*)(smem + 65536);     // [256][128]  (phase B, reuses wq)
  __bf16* q_lds  = (__bf16*)(smem + 131072);    // [128][256]
  __bf16* sc_lds = (__bf16*)(smem + 196608);    // [8][16][32]

  const int t = threadIdx.x, w = t >> 5, lane = t & 31, c = lane & 15, hi = lane >> 4;
  const int b = blockIdx.y;
  const int nw = blockIdx.x * 128 + w * 16;  // this wave's 16 rows

  // ---- stage Wq (bf16, 128KB = 256 rows x 64 8B-units) into LDS
#if defined(USE_TDM)
  if (w == 0) {
    tdm_load_2d(lds_off(wq_lds), Wqb, 64, 256, 64);
    __builtin_amdgcn_s_wait_tensorcnt(0);
  }
#elif defined(USE_ASYNC)
  {
#pragma unroll
    for (int i = 0; i < 32; ++i) {
      int off = (i * 256 + t) * 16;
      async_cp16(smem + off, (const char*)Wqb + off);
    }
    wait_async0();
  }
#else
  {
    const uint4* src = (const uint4*)Wqb;
    uint4* dst = (uint4*)smem;
#pragma unroll
    for (int i = 0; i < 32; ++i) dst[i * 256 + t] = src[i * 256 + t];
  }
#endif
  // prefetch residual rows needed by the epilogue
#pragma unroll
  for (int i = 0; i < 8; ++i)
    __builtin_prefetch(image + ((size_t)b * N_ + nw + i + 8 * hi) * D_ + c * 16, 0, 1);
  __syncthreads();

  // ---- Phase A: q = image @ Wq^T + bq
  const float* arow = image + ((size_t)b * N_ + nw + c) * D_;
  Frag afr[8];
#pragma unroll
  for (int kk = 0; kk < 8; ++kk) afr[kk] = load_frag_f32(arow, kk * 32, hi);

#pragma unroll
  for (int ct = 0; ct < 16; ++ct) {
    Acc acc;
    zero_acc(acc);
    const __bf16* brow = wq_lds + (ct * 16 + c) * D_;
#pragma unroll
    for (int kk = 0; kk < 8; ++kk) {
      Frag bf = load_frag_bf16(brow, kk * 32, hi);
      acc.v = wmma_bf16(afr[kk].v, bf.v, acc.v);
    }
    float bias = bq[ct * 16 + c];
#pragma unroll
    for (int i = 0; i < 8; ++i)  // D-layout scatter: (row i+8*hi, col c)
      q_lds[(w * 16 + i + 8 * hi) * D_ + ct * 16 + c] = (__bf16)(acc.e[i] + bias);
  }
  // reload q as A-fragments (layout conversion via LDS; same-wave DS is in-order)
  Frag qfr[8];
  const __bf16* qrow = q_lds + (w * 16 + c) * D_;
#pragma unroll
  for (int kk = 0; kk < 8; ++kk) qfr[kk] = load_frag_bf16(qrow, kk * 32, hi);

  __syncthreads();  // all waves done reading wq_lds before k/vT overwrite it

  Acc oacc[16];
#pragma unroll
  for (int dt = 0; dt < 16; ++dt) zero_acc(oacc[dt]);

  // ---- Phase B: two m-chunks of 128
  for (int ch = 0; ch < 2; ++ch) {
    // stage k chunk [128][256]bf16 (rows of 64 8B-units) and vT chunk [256][128]bf16
#if defined(USE_TDM)
    if (w == 0) {
      tdm_load_2d(lds_off(k_lds), kbf + ((size_t)b * M_ + ch * 128) * D_, 64, 128, 64);
      __builtin_amdgcn_s_wait_tensorcnt(0);
    } else if (w == 1) {
      tdm_load_2d(lds_off(vt_lds), vtb + (size_t)b * D_ * M_ + ch * 128, 32, 256, 64);
      __builtin_amdgcn_s_wait_tensorcnt(0);
    }
#elif defined(USE_ASYNC)
    {
      const char* ks = (const char*)(kbf + ((size_t)b * M_ + ch * 128) * D_);
#pragma unroll
      for (int i = 0; i < 16; ++i) {
        int off = (i * 256 + t) * 16;
        async_cp16((char*)k_lds + off, ks + off);
      }
      const __bf16* vs = vtb + (size_t)b * D_ * M_ + ch * 128;
      int rr = t >> 4, c4 = t & 15;
#pragma unroll
      for (int i = 0; i < 16; ++i) {
        int row = i * 16 + rr;
        async_cp16((char*)vt_lds + (row * 16 + c4) * 16,
                   (const char*)(vs + (size_t)row * M_ + c4 * 8));
      }
      wait_async0();
    }
#else
    {
      const uint4* ks = (const uint4*)(kbf + ((size_t)b * M_ + ch * 128) * D_);
      uint4* kd = (uint4*)k_lds;
#pragma unroll
      for (int i = 0; i < 16; ++i) kd[i * 256 + t] = ks[i * 256 + t];
      const __bf16* vs = vtb + (size_t)b * D_ * M_ + ch * 128;
      uint4* vd = (uint4*)vt_lds;
      int rr = t >> 4, c4 = t & 15;
#pragma unroll
      for (int i = 0; i < 16; ++i) {
        int row = i * 16 + rr;
        vd[row * 16 + c4] = *(const uint4*)(vs + (size_t)row * M_ + c4 * 8);
      }
    }
#endif
    __syncthreads();

#pragma unroll
    for (int mp = 0; mp < 4; ++mp) {  // 32-wide m strip
      Acc s0, s1;
      zero_acc(s0); zero_acc(s1);
      const __bf16* kr0 = k_lds + (mp * 32 + c) * D_;
      const __bf16* kr1 = k_lds + (mp * 32 + 16 + c) * D_;
#pragma unroll
      for (int kk = 0; kk < 8; ++kk) {
        Frag b0 = load_frag_bf16(kr0, kk * 32, hi);
        s0.v = wmma_bf16(qfr[kk].v, b0.v, s0.v);
        Frag b1 = load_frag_bf16(kr1, kk * 32, hi);
        s1.v = wmma_bf16(qfr[kk].v, b1.v, s1.v);
      }
      // sigmoid(score * 1/sqrt(256)); D-layout -> A-layout via tiny LDS round trip
      __bf16* sc = sc_lds + w * 16 * 32;
#pragma unroll
      for (int i = 0; i < 8; ++i) {
        float g0 = 1.0f / (1.0f + __expf(-0.0625f * s0.e[i]));
        float g1 = 1.0f / (1.0f + __expf(-0.0625f * s1.e[i]));
        sc[(i + 8 * hi) * 32 + c] = (__bf16)g0;
        sc[(i + 8 * hi) * 32 + 16 + c] = (__bf16)g1;
      }
      Frag sfr = load_frag_bf16(sc + c * 32, 0, hi);  // A-frag of sigmoid strip
#pragma unroll
      for (int dt = 0; dt < 16; ++dt) {
        const __bf16* vrow = vt_lds + (dt * 16 + c) * 128;
        Frag bv_ = load_frag_bf16(vrow, mp * 32, hi);
        oacc[dt].v = wmma_bf16(sfr.v, bv_.v, oacc[dt].v);
      }
    }
    __syncthreads();
  }

  // ---- epilogue: residual add + store
#pragma unroll
  for (int dt = 0; dt < 16; ++dt) {
#pragma unroll
    for (int i = 0; i < 8; ++i) {
      size_t idx = ((size_t)b * N_ + nw + i + 8 * hi) * D_ + dt * 16 + c;
      out[idx] = oacc[dt].e[i] + image[idx];
    }
  }
}

extern "C" void kernel_launch(void* const* d_in, const int* in_sizes, int n_in,
                              void* d_out, int out_size, void* d_ws, size_t ws_size,
                              hipStream_t stream) {
  const float* task  = (const float*)d_in[0];
  const float* image = (const float*)d_in[1];
  const float* Wq    = (const float*)d_in[2];
  const float* bq    = (const float*)d_in[3];
  const float* Wk    = (const float*)d_in[4];
  const float* bk    = (const float*)d_in[5];
  const float* Wv    = (const float*)d_in[6];
  const float* bv    = (const float*)d_in[7];
  float* out = (float*)d_out;

  // workspace: kbf (2MB) | vT (2MB) | Wq,Wk,Wv bf16 (128KB each)
  char* ws = (char*)d_ws;
  __bf16* kbf = (__bf16*)ws;
  __bf16* vtb = (__bf16*)(ws + (size_t)B_ * M_ * D_ * 2);
  __bf16* wqb = (__bf16*)(ws + (size_t)2 * B_ * M_ * D_ * 2);
  __bf16* wkb = wqb + (size_t)D_ * D_;
  __bf16* wvb = wkb + (size_t)D_ * D_;

  wcvt_kernel<<<(D_ * D_) / (256 * 4), 256, 0, stream>>>(Wq, wqb);
  wcvt_kernel<<<(D_ * D_) / (256 * 4), 256, 0, stream>>>(Wk, wkb);
  wcvt_kernel<<<(D_ * D_) / (256 * 4), 256, 0, stream>>>(Wv, wvb);
  project_kv_kernel<<<B_ * (M_ / 128), 256, 0, stream>>>(task, wkb, bk, wvb, bv, kbf, vtb);
  fused_kernel<<<dim3(N_ / 128, B_), 256, 0, stream>>>(image, wqb, bq, kbf, vtb, out);
}
